// MoEPolicy_77378130804783
// MI455X (gfx1250) — matrix operand order repaired and checked by reading, not compiled, not run on hardware
//
#include <hip/hip_runtime.h>
#include <cstdint>
#include <cstddef>

#define DD     128
#define TDIM   128
#define NTOK   64
#define NEXP   16
#define KSH    2
#define NB     32
#define MNODE  10000
#define NNODE  20000
#define NEDGE  160000
#define H4     512
#define NTILES 1250   // 20000 / 16 exactly

typedef __bf16 bf16_t;
typedef __attribute__((ext_vector_type(16))) __bf16 v16bf;
typedef __attribute__((ext_vector_type(8)))  __bf16 v8bf;
typedef __attribute__((ext_vector_type(4)))  __bf16 v4bf;
typedef __attribute__((ext_vector_type(8)))  float  v8f;

// ---------------- WMMA helpers (CDNA5 16x16x32 bf16, f32 accum) -------------

__device__ __forceinline__ v8f wmma_bf16(v16bf a, v16bf b, v8f c) {
  // 8 args: (neg_a, A, neg_b, B, c_mod, C, reuse_a, reuse_b)
  return __builtin_amdgcn_wmma_f32_16x16x32_bf16(false, a, false, b, (short)0, c,
                                                 false, false);
}

// A fragment from a row-major 16x32 bf16 tile (global or LDS), leading dim `ld`.
// ISA 7.12.2: lane (m = lane&15, g = lane>>4) holds K = g*8..g*8+7 then
// 16+g*8..16+g*8+7 -> two contiguous 16B runs -> two b128 loads.
__device__ __forceinline__ v16bf loadA(const bf16_t* base, int ld) {
  int lane = threadIdx.x & 31;
  int m = lane & 15, g = lane >> 4;
  v8bf lo = *(const v8bf*)(base + m * ld + g * 8);
  v8bf hi = *(const v8bf*)(base + m * ld + 16 + g * 8);
  return __builtin_shufflevector(lo, hi, 0, 1, 2, 3, 4, 5, 6, 7,
                                 8, 9, 10, 11, 12, 13, 14, 15);
}

// B fragment from pre-swizzled weights: each 32x16 (KxN) chunk stored as
// [lane][16] contiguous bf16 (lane: n = lane&15, g = lane>>4; elem i -> K=(i>>3)*16+g*8+(i&7)).
__device__ __forceinline__ v16bf loadBswz(const bf16_t* chunk) {
  int lane = threadIdx.x & 31;
  const v8bf* p = (const v8bf*)(chunk + lane * 16);
  v8bf lo = p[0], hi = p[1];
  return __builtin_shufflevector(lo, hi, 0, 1, 2, 3, 4, 5, 6, 7,
                                 8, 9, 10, 11, 12, 13, 14, 15);
}

__device__ __forceinline__ float gelu_exact(float x) {
  return 0.5f * x * (1.0f + erff(x * 0.70710678118654752f));
}

// ---------------- weight conversion + fragment swizzle ----------------------
// src: nmat row-major [Kt x Nt] f32 matrices. dst: per matrix, chunks ordered
// [j][kc] (ck = Kt/32 chunks per j), each chunk = 32 lanes * 16 bf16.

__global__ __launch_bounds__(256) void k_swz(const float* src, bf16_t* dst,
                                             int nmat, int Kt, int Nt) {
  int i = blockIdx.x * 256 + threadIdx.x;
  int per = Kt * Nt;
  if (i >= nmat * per) return;
  int m = i / per, r = i % per;
  int chunk = r >> 9;            // 512 elems per chunk
  int lane = (r >> 4) & 31, e = r & 15;
  int ck = Kt >> 5;
  int j = chunk / ck, kc = chunk % ck;
  int g = lane >> 4, n = lane & 15;
  int k = kc * 32 + (e >> 3) * 16 + g * 8 + (e & 7);
  dst[i] = (bf16_t)src[(size_t)m * per + (size_t)k * Nt + (j * 16 + n)];
}

// Transposed source: B[k][n] = src[n*Kt + k]  (for tokK: scores = Q @ tokK^T)
__global__ __launch_bounds__(256) void k_swzT(const float* src, bf16_t* dst,
                                              int Kt, int Nt) {
  int i = blockIdx.x * 256 + threadIdx.x;
  if (i >= Kt * Nt) return;
  int chunk = i >> 9;
  int lane = (i >> 4) & 31, e = i & 15;
  int ck = Kt >> 5;
  int j = chunk / ck, kc = chunk % ck;
  int g = lane >> 4, n = lane & 15;
  int k = kc * 32 + (e >> 3) * 16 + g * 8 + (e & 7);
  dst[i] = (bf16_t)src[(size_t)(j * 16 + n) * Kt + k];
}

// ---------------- encoder ----------------------------------------------------

__global__ __launch_bounds__(256) void k_cemb(const float* cf, const float* Wc,
                                              const float* bc, float* cemb) {
  int i = blockIdx.x * 256 + threadIdx.x;   // exact: MNODE*DD % 256 == 0
  int r = i >> 7, c = i & 127;
  float s = bc[c];
#pragma unroll
  for (int k = 0; k < 4; ++k) s += cf[r * 4 + k] * Wc[k * DD + c];
  cemb[i] = s;
}

__global__ __launch_bounds__(256) void k_v0(const float* vf, const float* Wv,
                                            const float* bv, float* vacc) {
  int i = blockIdx.x * 256 + threadIdx.x;   // exact: NNODE*DD % 256 == 0
  int r = i >> 7, c = i & 127;
  float s = bv[c];
#pragma unroll
  for (int k = 0; k < 6; ++k) s += vf[r * 6 + k] * Wv[k * DD + c];
  vacc[i] = s;
}

__global__ __launch_bounds__(256) void k_edges(const float* cemb, const float* ea,
                                               const int* econs, const int* evars,
                                               const float* We, const float* be,
                                               float* vacc) {
  int t = blockIdx.x * 256 + threadIdx.x;   // exact: NEDGE*32 % 256 == 0
  int e = t >> 5, q = t & 31;
  int c0 = q * 4;
  float a = ea[e];
  int ci = econs[e], vi = evars[e];
  float4 wv = *(const float4*)(We + c0);
  float4 bv = *(const float4*)(be + c0);
  float4 cv = *(const float4*)(cemb + (size_t)ci * DD + c0);
  float* dst = vacc + (size_t)vi * DD + c0;
  atomicAdd(dst + 0, cv.x * (a * wv.x + bv.x));
  atomicAdd(dst + 1, cv.y * (a * wv.y + bv.y));
  atomicAdd(dst + 2, cv.z * (a * wv.z + bv.z));
  atomicAdd(dst + 3, cv.w * (a * wv.w + bv.w));
}

// relu + LayerNorm per row (one wave per row): f32 residual, packed bf16 copy,
// graph-pool partial sums.
__global__ __launch_bounds__(128) void k_lnv(const float* vacc, const float* ln_g,
                                             const float* ln_b, const int* bidx,
                                             bf16_t* vembbf, float* combined,
                                             float* gsum, float* cnt) {
  int row = blockIdx.x * 4 + (threadIdx.x >> 5);   // 5000 blocks * 4 = NNODE
  int lane = threadIdx.x & 31;
  float4 x4 = ((const float4*)(vacc + (size_t)row * DD))[lane];
  float v[4] = {fmaxf(x4.x, 0.f), fmaxf(x4.y, 0.f), fmaxf(x4.z, 0.f), fmaxf(x4.w, 0.f)};
  float s = v[0] + v[1] + v[2] + v[3];
#pragma unroll
  for (int o = 16; o > 0; o >>= 1) s += __shfl_xor(s, o, 32);
  float mu = s * (1.f / 128.f);
  float d0 = v[0] - mu, d1 = v[1] - mu, d2 = v[2] - mu, d3 = v[3] - mu;
  float q = d0 * d0 + d1 * d1 + d2 * d2 + d3 * d3;
#pragma unroll
  for (int o = 16; o > 0; o >>= 1) q += __shfl_xor(q, o, 32);
  float rstd = rsqrtf(q * (1.f / 128.f) + 1e-5f);
  int b = bidx[row];
  float y[4];
#pragma unroll
  for (int i = 0; i < 4; ++i) {
    int c = lane * 4 + i;
    y[i] = (v[i] - mu) * rstd * ln_g[c] + ln_b[c];
    atomicAdd(&gsum[b * DD + c], y[i]);
  }
  v4bf h = {(bf16_t)y[0], (bf16_t)y[1], (bf16_t)y[2], (bf16_t)y[3]};
  *(v4bf*)(vembbf + (size_t)row * DD + lane * 4) = h;
  float4 o4 = {y[0], y[1], y[2], y[3]};
  ((float4*)(combined + (size_t)row * DD))[lane] = o4;
  if (lane == 0) atomicAdd(&cnt[b], 1.f);
}

// ---------------- struct-token memory (fused 3-GEMM + softmax) ---------------

__global__ __launch_bounds__(32) void k_struct(const bf16_t* vembbf, const bf16_t* wq_s,
                                               const bf16_t* tokKT_s, const bf16_t* tokV_s,
                                               const float* bq, const int* bidx,
                                               float* ssum) {
  __shared__ alignas(16) bf16_t Qbf[16 * 128];
  __shared__ float Sf[16 * 64];
  __shared__ alignas(16) bf16_t Wts[16 * 64];
  __shared__ int bidx_s[16];

  int tile = blockIdx.x;
  int lane = threadIdx.x;
  int n0 = tile * 16;
  int nn = lane & 15, g2 = lane >> 4;

  if (lane < 16) bidx_s[lane] = bidx[n0 + lane];

  // A fragments straight from global row-major bf16 activations
  v16bf a[4];
#pragma unroll
  for (int kc = 0; kc < 4; ++kc)
    a[kc] = loadA(vembbf + (size_t)n0 * DD + kc * 32, DD);

  // Q = X @ Wq + bq
#pragma unroll
  for (int j = 0; j < 8; ++j) {
    v8f acc = {};
#pragma unroll
    for (int kc = 0; kc < 4; ++kc)
      acc = wmma_bf16(a[kc], loadBswz(wq_s + (size_t)(j * 4 + kc) * 512), acc);
    float bqv = bq[j * 16 + nn];
#pragma unroll
    for (int jj = 0; jj < 8; ++jj)
      Qbf[(g2 * 8 + jj) * 128 + j * 16 + nn] = (bf16_t)(acc[jj] + bqv);
  }
  __syncthreads();

  v16bf aq[4];
#pragma unroll
  for (int kc = 0; kc < 4; ++kc) aq[kc] = loadA(Qbf + kc * 32, 128);

  // scores = Q @ tokK^T / sqrt(TD)
#pragma unroll
  for (int j = 0; j < 4; ++j) {
    v8f acc = {};
#pragma unroll
    for (int kc = 0; kc < 4; ++kc)
      acc = wmma_bf16(aq[kc], loadBswz(tokKT_s + (size_t)(j * 4 + kc) * 512), acc);
#pragma unroll
    for (int jj = 0; jj < 8; ++jj)
      Sf[(g2 * 8 + jj) * 64 + j * 16 + nn] = acc[jj] * 0.088388347648318447f;
  }
  __syncthreads();

  // row softmax over 64 tokens
  if (lane < 16) {
    float mx = -1e30f;
    for (int c = 0; c < 64; ++c) mx = fmaxf(mx, Sf[lane * 64 + c]);
    float s = 0.f;
    for (int c = 0; c < 64; ++c) s += expf(Sf[lane * 64 + c] - mx);
    float inv = 1.f / s;
    for (int c = 0; c < 64; ++c)
      Wts[lane * 64 + c] = (bf16_t)(expf(Sf[lane * 64 + c] - mx) * inv);
  }
  __syncthreads();

  v16bf aw[2];
#pragma unroll
  for (int kc = 0; kc < 2; ++kc) aw[kc] = loadA(Wts + kc * 32, 64);

  // node_struct = wts @ tokV ; pool into ssum
#pragma unroll
  for (int j = 0; j < 8; ++j) {
    v8f acc = {};
#pragma unroll
    for (int kc = 0; kc < 2; ++kc)
      acc = wmma_bf16(aw[kc], loadBswz(tokV_s + (size_t)(j * 2 + kc) * 512), acc);
#pragma unroll
    for (int jj = 0; jj < 8; ++jj) {
      int b = bidx_s[g2 * 8 + jj];
      atomicAdd(&ssum[b * TDIM + j * 16 + nn], acc[jj]);
    }
  }
}

// ---------------- gating (B=32, tiny) ---------------------------------------

__global__ __launch_bounds__(32) void k_gate(const float* gsum, const float* ssum,
                                             const float* cnt, const float* Wg,
                                             const float* bg, const float* alpha,
                                             const float* ebias, float* route) {
  int g = threadIdx.x;
  float c = cnt[g];
  float inv = 1.f / fmaxf(c, 1.f);
  float acc[NEXP];
#pragma unroll
  for (int e = 0; e < NEXP; ++e) acc[e] = bg[e];
  for (int k = 0; k < 256; ++k) {
    float x = (k < 128 ? gsum[g * DD + k] : ssum[g * TDIM + (k - 128)]) * inv;
#pragma unroll
    for (int e = 0; e < NEXP; ++e) acc[e] += x * Wg[k * NEXP + e];
  }
  float sc = alpha[0] / 0.6f;
  float lg[NEXP];
#pragma unroll
  for (int e = 0; e < NEXP; ++e) lg[e] = acc[e] * sc + ebias[e];
  bool pick[NEXP];
#pragma unroll
  for (int e = 0; e < NEXP; ++e) pick[e] = false;
  for (int t = 0; t < 4; ++t) {
    float best = -1e30f; int bi = 0;
    for (int e = 0; e < NEXP; ++e)
      if (!pick[e] && lg[e] > best) { best = lg[e]; bi = e; }
    pick[bi] = true;
  }
  float mx = lg[0];
  for (int e = 1; e < NEXP; ++e) mx = fmaxf(mx, lg[e]);
  float ex[NEXP], s = 0.f;
  for (int e = 0; e < NEXP; ++e) { ex[e] = expf(lg[e] - mx); s += ex[e]; }
  float ws = 0.f;
  for (int e = 0; e < NEXP; ++e) { ex[e] = pick[e] ? ex[e] / s : 0.f; ws += ex[e]; }
  float rinv = 1.f / (ws + 1e-12f);
  for (int e = 0; e < NEXP; ++e) route[g * NEXP + e] = ex[e] * rinv;
}

// ---------------- expert FFN (the big WMMA kernel) ---------------------------
// grid = (1250 tiles, 18 slots): slots 0..15 dedicated (skip if route==0),
// slots 16..17 shared (weight 0.5 each = mean of KS=2).

__global__ __launch_bounds__(128) void k_experts(
    const bf16_t* vembbf, const bf16_t* dW1s, const bf16_t* dW2s,
    const bf16_t* sW1s, const bf16_t* sW2s,
    const float* db1, const float* db2, const float* dg, const float* dbb,
    const float* sb1, const float* sb2, const float* sg, const float* sbb,
    const float* route, const int* bidx, float* combined) {
  __shared__ alignas(16) bf16_t Hbf[16 * 512];
  __shared__ float Of[16 * 128];
  __shared__ float wrow[16];
  __shared__ float mu_s[16], rs_s[16];
  __shared__ int anyact;

  int tile = blockIdx.x, slot = blockIdx.y;
  int tid = threadIdx.x, w = tid >> 5, lane = tid & 31;
  int nn = lane & 15, g2 = lane >> 4;
  int n0 = tile * 16;

  const bf16_t *W1s, *W2s;
  const float *b1p, *b2p, *gp, *bbp;
  if (slot < NEXP) {
    int e = slot;
    W1s = dW1s + (size_t)e * DD * H4;  W2s = dW2s + (size_t)e * H4 * DD;
    b1p = db1 + e * H4;  b2p = db2 + e * DD;  gp = dg + e * DD;  bbp = dbb + e * DD;
  } else {
    int e = slot - NEXP;
    W1s = sW1s + (size_t)e * DD * H4;  W2s = sW2s + (size_t)e * H4 * DD;
    b1p = sb1 + e * H4;  b2p = sb2 + e * DD;  gp = sg + e * DD;  bbp = sbb + e * DD;
  }

  if (tid < 16) {
    int b = bidx[n0 + tid];
    wrow[tid] = (slot < NEXP) ? route[b * NEXP + slot] : 0.5f;
  }
  if (tid == 0) anyact = 0;
  __syncthreads();
  if (tid == 0) {
    int a = 0;
    for (int r = 0; r < 16; ++r) if (wrow[r] != 0.f) a = 1;
    anyact = a;
  }
  __syncthreads();
  if (!anyact) return;   // uniform block exit: inactive dedicated expert

  __builtin_prefetch(W1s + (size_t)w * 8 * 4 * 512, 0, 3);
  __builtin_prefetch(W2s + (size_t)w * 2 * 16 * 512, 0, 3);

  v16bf a[4];
#pragma unroll
  for (int kc = 0; kc < 4; ++kc)
    a[kc] = loadA(vembbf + (size_t)n0 * DD + kc * 32, DD);

  // ---- stage 1: H = gelu(X @ W1 + b1) -> bf16 LDS. Wave w: N-tiles w*8..w*8+7
#pragma unroll
  for (int jt = 0; jt < 8; ++jt) {
    int j = w * 8 + jt;
    v8f acc = {};
#pragma unroll
    for (int kc = 0; kc < 4; ++kc)
      acc = wmma_bf16(a[kc], loadBswz(W1s + (size_t)(j * 4 + kc) * 512), acc);
    float b1v = b1p[j * 16 + nn];
#pragma unroll
    for (int jj = 0; jj < 8; ++jj)
      Hbf[(g2 * 8 + jj) * 512 + j * 16 + nn] = (bf16_t)gelu_exact(acc[jj] + b1v);
  }
  __syncthreads();

  // ---- stage 2: O = H @ W2 + b2 (K = 512). Wave w: N-tiles w*2, w*2+1
  {
    int j0 = w * 2, j1 = w * 2 + 1;
    v8f acc0 = {}, acc1 = {};
#pragma unroll
    for (int kc = 0; kc < 16; ++kc) {
      v16bf a2 = loadA(Hbf + kc * 32, 512);
      acc0 = wmma_bf16(a2, loadBswz(W2s + (size_t)(j0 * 16 + kc) * 512), acc0);
      acc1 = wmma_bf16(a2, loadBswz(W2s + (size_t)(j1 * 16 + kc) * 512), acc1);
    }
    float bv0 = b2p[j0 * 16 + nn], bv1 = b2p[j1 * 16 + nn];
#pragma unroll
    for (int jj = 0; jj < 8; ++jj) {
      Of[(g2 * 8 + jj) * 128 + j0 * 16 + nn] = acc0[jj] + bv0;
      Of[(g2 * 8 + jj) * 128 + j1 * 16 + nn] = acc1[jj] + bv1;
    }
  }
  __syncthreads();

  // ---- per-row LayerNorm stats
  if (tid < 16) {
    float s = 0.f;
    for (int c = 0; c < 128; ++c) s += Of[tid * 128 + c];
    float mu = s * (1.f / 128.f);
    float q = 0.f;
    for (int c = 0; c < 128; ++c) { float d = Of[tid * 128 + c] - mu; q += d * d; }
    mu_s[tid] = mu;
    rs_s[tid] = rsqrtf(q * (1.f / 128.f) + 1e-5f);
  }
  __syncthreads();

  // ---- weighted accumulate into combined
  {
    int r = tid >> 3, c0 = (tid & 7) * 16;
    float wv = wrow[r];
    if (wv != 0.f) {
      float mu = mu_s[r], rs = rs_s[r];
      size_t node = (size_t)(n0 + r);
#pragma unroll
      for (int i = 0; i < 16; ++i) {
        int c = c0 + i;
        float val = (Of[r * 128 + c] - mu) * rs * gp[c] + bbp[c];
        atomicAdd(&combined[node * DD + c], wv * val);
      }
    }
  }
}

// ---------------- task head --------------------------------------------------

__global__ __launch_bounds__(128) void k_head(const float* combined, const bf16_t* hW1s,
                                              const float* hb1, const float* hW2,
                                              const float* hb2, float* out) {
  __shared__ alignas(16) bf16_t Xbf[16 * 128];
  __shared__ float Yf[16 * 128];
  __shared__ float part[16 * 8];

  int tile = blockIdx.x, tid = threadIdx.x, w = tid >> 5, lane = tid & 31;
  int nn = lane & 15, g2 = lane >> 4;
  int n0 = tile * 16;

  {
    int r = tid >> 3, q = tid & 7;
    const float4* s4 = (const float4*)(combined + (size_t)(n0 + r) * DD + q * 16);
#pragma unroll
    for (int t = 0; t < 4; ++t) {
      float4 f = s4[t];
      v4bf h = {(bf16_t)f.x, (bf16_t)f.y, (bf16_t)f.z, (bf16_t)f.w};
      *(v4bf*)(Xbf + r * 128 + q * 16 + t * 4) = h;
    }
  }
  __syncthreads();

  v16bf a[4];
#pragma unroll
  for (int kc = 0; kc < 4; ++kc) a[kc] = loadA(Xbf + kc * 32, 128);

#pragma unroll
  for (int jt = 0; jt < 2; ++jt) {
    int j = w * 2 + jt;
    v8f acc = {};
#pragma unroll
    for (int kc = 0; kc < 4; ++kc)
      acc = wmma_bf16(a[kc], loadBswz(hW1s + (size_t)(j * 4 + kc) * 512), acc);
    float bv = hb1[j * 16 + nn];
#pragma unroll
    for (int jj = 0; jj < 8; ++jj)
      Yf[(g2 * 8 + jj) * 128 + j * 16 + nn] = fmaxf(acc[jj] + bv, 0.f);
  }
  __syncthreads();

  {
    int r = tid >> 3, q = tid & 7;
    float s = 0.f;
#pragma unroll
    for (int i = 0; i < 16; ++i) s += Yf[r * 128 + q * 16 + i] * hW2[q * 16 + i];
    part[r * 8 + q] = s;
  }
  __syncthreads();
  if (tid < 16) {
    float s = hb2[0];
#pragma unroll
    for (int i = 0; i < 8; ++i) s += part[tid * 8 + i];
    out[n0 + tid] = s;
  }
}

// ---------------- launch -----------------------------------------------------

extern "C" void kernel_launch(void* const* d_in, const int* in_sizes, int n_in,
                              void* d_out, int out_size, void* d_ws, size_t ws_size,
                              hipStream_t stream) {
  (void)in_sizes; (void)n_in; (void)out_size; (void)ws_size;
  const float* c_feat   = (const float*)d_in[0];
  const float* edge_attr= (const float*)d_in[1];
  const float* v_feat   = (const float*)d_in[2];
  const int*   e_cons   = (const int*)d_in[3];
  const int*   e_vars   = (const int*)d_in[4];
  const int*   bidx     = (const int*)d_in[5];
  const float* Wc = (const float*)d_in[6];  const float* bc = (const float*)d_in[7];
  const float* Wv = (const float*)d_in[8];  const float* bv = (const float*)d_in[9];
  const float* We = (const float*)d_in[10]; const float* be = (const float*)d_in[11];
  const float* ln_g = (const float*)d_in[12]; const float* ln_b = (const float*)d_in[13];
  const float* Wq = (const float*)d_in[14]; const float* bq = (const float*)d_in[15];
  const float* tokK = (const float*)d_in[16]; const float* tokV = (const float*)d_in[17];
  const float* Wg = (const float*)d_in[18]; const float* bg = (const float*)d_in[19];
  const float* alpha = (const float*)d_in[20]; const float* ebias = (const float*)d_in[21];
  const float* sW1 = (const float*)d_in[22]; const float* sb1 = (const float*)d_in[23];
  const float* sW2 = (const float*)d_in[24]; const float* sb2 = (const float*)d_in[25];
  const float* sg  = (const float*)d_in[26]; const float* sbb = (const float*)d_in[27];
  const float* dW1 = (const float*)d_in[28]; const float* db1 = (const float*)d_in[29];
  const float* dW2 = (const float*)d_in[30]; const float* db2 = (const float*)d_in[31];
  const float* dg  = (const float*)d_in[32]; const float* dbb = (const float*)d_in[33];
  const float* hW1 = (const float*)d_in[34]; const float* hb1 = (const float*)d_in[35];
  const float* hW2 = (const float*)d_in[36]; const float* hb2 = (const float*)d_in[37];
  float* out = (float*)d_out;

  uint8_t* ws = (uint8_t*)d_ws;
  size_t off = 0;
  auto carve = [&](size_t bytes) -> void* {
    void* p = ws + off;
    off = (off + bytes + 255) & ~(size_t)255;
    return p;
  };
  float*  cemb     = (float*)carve((size_t)MNODE * DD * 4);
  float*  vacc     = (float*)carve((size_t)NNODE * DD * 4);
  bf16_t* vembbf   = (bf16_t*)carve((size_t)NNODE * DD * 2);
  float*  combined = (float*)carve((size_t)NNODE * DD * 4);
  float*  gsum     = (float*)carve((size_t)NB * DD * 4);
  float*  ssum     = (float*)carve((size_t)NB * TDIM * 4);
  float*  cnt      = (float*)carve((size_t)NB * 4);
  float*  route    = (float*)carve((size_t)NB * NEXP * 4);
  bf16_t* wq_s     = (bf16_t*)carve((size_t)DD * TDIM * 2);
  bf16_t* tokKT_s  = (bf16_t*)carve((size_t)TDIM * NTOK * 2);
  bf16_t* tokV_s   = (bf16_t*)carve((size_t)NTOK * TDIM * 2);
  bf16_t* sW1s     = (bf16_t*)carve((size_t)KSH * DD * H4 * 2);
  bf16_t* sW2s     = (bf16_t*)carve((size_t)KSH * H4 * DD * 2);
  bf16_t* dW1s     = (bf16_t*)carve((size_t)NEXP * DD * H4 * 2);
  bf16_t* dW2s     = (bf16_t*)carve((size_t)NEXP * H4 * DD * 2);
  bf16_t* hW1s     = (bf16_t*)carve((size_t)DD * DD * 2);

  hipMemsetAsync(gsum, 0, (size_t)NB * DD * 4, stream);
  hipMemsetAsync(ssum, 0, (size_t)NB * TDIM * 4, stream);
  hipMemsetAsync(cnt, 0, (size_t)NB * 4, stream);

  auto SWZ = [&](const float* s, bf16_t* d, int nmat, int Kt, int Nt) {
    int n = nmat * Kt * Nt;
    k_swz<<<(n + 255) / 256, 256, 0, stream>>>(s, d, nmat, Kt, Nt);
  };
  SWZ(Wq, wq_s, 1, DD, TDIM);
  k_swzT<<<(TDIM * NTOK + 255) / 256, 256, 0, stream>>>(tokK, tokKT_s, TDIM, NTOK);
  SWZ(tokV, tokV_s, 1, NTOK, TDIM);
  SWZ(sW1, sW1s, KSH, DD, H4);
  SWZ(sW2, sW2s, KSH, H4, DD);
  SWZ(dW1, dW1s, NEXP, DD, H4);
  SWZ(dW2, dW2s, NEXP, H4, DD);
  SWZ(hW1, hW1s, 1, DD, DD);

  k_cemb<<<(MNODE * DD) / 256, 256, 0, stream>>>(c_feat, Wc, bc, cemb);
  k_v0<<<(NNODE * DD) / 256, 256, 0, stream>>>(v_feat, Wv, bv, vacc);
  k_edges<<<(NEDGE * 32) / 256, 256, 0, stream>>>(cemb, edge_attr, e_cons, e_vars, We, be, vacc);
  k_lnv<<<NNODE / 4, 128, 0, stream>>>(vacc, ln_g, ln_b, bidx, vembbf, combined, gsum, cnt);
  k_struct<<<NTILES, 32, 0, stream>>>(vembbf, wq_s, tokKT_s, tokV_s, bq, bidx, ssum);
  k_gate<<<1, 32, 0, stream>>>(gsum, ssum, cnt, Wg, bg, alpha, ebias, route);
  k_experts<<<dim3(NTILES, NEXP + KSH), 128, 0, stream>>>(
      vembbf, dW1s, dW2s, sW1s, sW2s, db1, db2, dg, dbb, sb1, sb2, sg, sbb,
      route, bidx, combined);
  k_head<<<NTILES, 128, 0, stream>>>(combined, hW1s, hb1, hW2, hb2, out);
}
